// DOMINOPlusPlusLoss_fast_2233382994513
// MI455X (gfx1250) — matrix operand order repaired and checked by reading, not compiled, not run on hardware
//
#include <hip/hip_runtime.h>
#include <hip/hip_bf16.h>

typedef __attribute__((ext_vector_type(16))) _Float16 v16h;
typedef __attribute__((ext_vector_type(8)))  float    v8f;

#define NCLS        12
#define SPATIAL     884736          // 96*96*96
#define VOXPB       4096            // voxels per block
#define BLKS_PER_N  216             // SPATIAL / VOXPB
#define NBLK        432             // 2 * BLKS_PER_N
#define SLOT        192             // floats per ws slot (144 M1 + 12 I + 12 G + 12 P + 1 CE, padded)
#define PCH         17              // padded channel stride in LDS (16 + 1, conflict-free)

__device__ __forceinline__ float wred32(float x) {
    #pragma unroll
    for (int o = 16; o > 0; o >>= 1) x += __shfl_down(x, o, 32);
    return x;
}

__global__ __launch_bounds__(256) void domino_main_kernel(
    const float* __restrict__ logits,      // (2, 12, S)
    const long long* __restrict__ target,  // (2, 1, S) int64
    float* __restrict__ ws)                // NBLK * SLOT floats
{
    __shared__ float sh_p[8 * 32 * PCH];   // per-wave probs tiles (32 voxels x 16 ch, padded)
    __shared__ int   sh_t[8 * 32];         // per-wave targets
    __shared__ float sh_m1[8 * 32 * 8];    // per-wave WMMA C dump
    __shared__ float sh_red[8 * 40];       // per-wave scalar partials

    const int tid  = threadIdx.x;
    const int wave = tid >> 5;
    const int lane = tid & 31;

    const int n   = (blockIdx.x >= BLKS_PER_N) ? 1 : 0;
    const int bin = blockIdx.x - n * BLKS_PER_N;
    const int vbase = bin * VOXPB + wave * 512;

    const float* lbase = logits + (size_t)n * NCLS * SPATIAL;
    const long long* tbase = target + (size_t)n * SPATIAL;

    float accI[NCLS], accG[NCLS], accP[NCLS];
    #pragma unroll
    for (int c = 0; c < NCLS; ++c) { accI[c] = 0.f; accG[c] = 0.f; accP[c] = 0.f; }
    float accCE = 0.f;
    v8f c_acc = {};

    float* pr = &sh_p[(wave * 32 + lane) * PCH];
    const float* pl = &sh_p[wave * 32 * PCH];
    const int*   tl = &sh_t[wave * 32];
    const int  mrow = lane & 15;
    const bool hi   = lane >= 16;
    const int kb0 = hi ? 8 : 0, kb1 = hi ? 24 : 16, kB = hi ? 16 : 0;

    #pragma unroll 1
    for (int it = 0; it < 16; ++it) {
        const int v = vbase + it * 32 + lane;

        // ---- softmax over 12 channels (coalesced strided loads) ----
        float l[NCLS];
        const float* lp = lbase + v;
        #pragma unroll
        for (int c = 0; c < NCLS; ++c) l[c] = lp[(size_t)c * SPATIAL];
        float mx = l[0];
        #pragma unroll
        for (int c = 1; c < NCLS; ++c) mx = fmaxf(mx, l[c]);
        float e[NCLS]; float s = 0.f;
        #pragma unroll
        for (int c = 0; c < NCLS; ++c) { e[c] = __expf(l[c] - mx); s += e[c]; }
        const float inv = 1.0f / s;
        const float lsum = __logf(s);

        const int tc = (int)tbase[v];
        float e_t = e[0], l_t = l[0];
        #pragma unroll
        for (int c = 1; c < NCLS; ++c) { const bool sel = (tc == c); e_t = sel ? e[c] : e_t; l_t = sel ? l[c] : l_t; }
        const float p_t = e_t * inv;
        accCE += (l_t - mx - lsum);

        #pragma unroll
        for (int c = 0; c < NCLS; ++c) {
            const float pc = e[c] * inv;
            accP[c] += pc;
            pr[c] = pc;
            const bool sel = (tc == c);
            accI[c] += sel ? p_t : 0.f;
            accG[c] += sel ? 1.f : 0.f;
        }
        pr[12] = 0.f; pr[13] = 0.f; pr[14] = 0.f; pr[15] = 0.f;
        sh_t[wave * 32 + lane] = tc;

        // LDS is in-order within a wave; fence the compiler + wait dscnt.
        asm volatile("s_wait_dscnt 0x0" ::: "memory");

        // ---- A: 16x32 onehot(target) f16, documented wave32 layout ----
        v16h a, b;
        #pragma unroll
        for (int i = 0; i < 8; ++i) a[i]     = (_Float16)((tl[kb0 + i] == mrow) ? 1.0f : 0.0f);
        #pragma unroll
        for (int i = 0; i < 8; ++i) a[8 + i] = (_Float16)((tl[kb1 + i] == mrow) ? 1.0f : 0.0f);
        // ---- B: 32x16 probs (K = voxel, N = channel), LDS transpose reads ----
        #pragma unroll
        for (int i = 0; i < 8; ++i) b[i]     = (_Float16)pl[(kb0 + i) * PCH + mrow];
        #pragma unroll
        for (int i = 0; i < 8; ++i) b[8 + i] = (_Float16)pl[(kb1 + i) * PCH + mrow];

        // M1 += onehot^T-style GEMM: D[m][c] = sum_k 1{t_k==m} * probs[k][c]
        c_acc = __builtin_amdgcn_wmma_f32_16x16x32_f16(false, a, false, b,
                                                       (short)0, c_acc, false, false);
    }

    // ---- dump per-wave WMMA accumulator ----
    #pragma unroll
    for (int j = 0; j < 8; ++j) sh_m1[wave * 256 + lane * 8 + j] = c_acc[j];

    // ---- wave-level reduction of scalar accumulators ----
    #pragma unroll
    for (int c = 0; c < NCLS; ++c) {
        float rI = wred32(accI[c]); if (lane == 0) sh_red[wave * 40 + c]      = rI;
        float rG = wred32(accG[c]); if (lane == 0) sh_red[wave * 40 + 12 + c] = rG;
        float rP = wred32(accP[c]); if (lane == 0) sh_red[wave * 40 + 24 + c] = rP;
    }
    float rC = wred32(accCE); if (lane == 0) sh_red[wave * 40 + 36] = rC;
    __syncthreads();

    // ---- block partials -> private ws slot (deterministic, no atomics) ----
    float* out = ws + (size_t)blockIdx.x * SLOT;
    if (tid < 144) {
        const int m = tid / 12, c = tid % 12;
        const int l = c + ((m >= 8) ? 16 : 0);
        const int j = m & 7;
        float sum = 0.f;
        #pragma unroll
        for (int w = 0; w < 8; ++w) sum += sh_m1[w * 256 + l * 8 + j];
        out[m * 12 + c] = sum;
    }
    if (tid < 37) {
        float sum = 0.f;
        #pragma unroll
        for (int w = 0; w < 8; ++w) sum += sh_red[w * 40 + tid];
        out[144 + tid] = sum;   // 144..155 inter, 156..167 ground, 168..179 pred, 180 ce
    }
}

__global__ __launch_bounds__(256) void domino_final_kernel(
    const float* __restrict__ ws,
    const float* __restrict__ pen_mat,     // (12,12)
    const int* __restrict__ gstep,
    const int* __restrict__ miter,
    float* __restrict__ out)
{
    __shared__ float sm[224];  // [0..143] M1 total, [144..167] inter(n,c), [168..191] ground, [192..215] pred, [216] ce
    const int tid = threadIdx.x;

    if (tid < 144) {
        float s = 0.f;
        for (int b = 0; b < NBLK; ++b) s += ws[(size_t)b * SLOT + tid];
        sm[tid] = s;
    } else if (tid < 168) {
        const int idx = tid - 144, nn = idx / 12, c = idx % 12;
        float s = 0.f;
        for (int b = nn * BLKS_PER_N; b < (nn + 1) * BLKS_PER_N; ++b) s += ws[(size_t)b * SLOT + 144 + c];
        sm[tid] = s;
    } else if (tid < 192) {
        const int idx = tid - 168, nn = idx / 12, c = idx % 12;
        float s = 0.f;
        for (int b = nn * BLKS_PER_N; b < (nn + 1) * BLKS_PER_N; ++b) s += ws[(size_t)b * SLOT + 156 + c];
        sm[tid] = s;
    } else if (tid < 216) {
        const int idx = tid - 192, nn = idx / 12, c = idx % 12;
        float s = 0.f;
        for (int b = nn * BLKS_PER_N; b < (nn + 1) * BLKS_PER_N; ++b) s += ws[(size_t)b * SLOT + 168 + c];
        sm[tid] = s;
    } else if (tid == 216) {
        float s = 0.f;
        for (int b = 0; b < NBLK; ++b) s += ws[(size_t)b * SLOT + 180];
        sm[216] = s;
    }
    __syncthreads();

    if (tid == 0) {
        const float TOT = 2.0f * (float)SPATIAL;
        const float ce = -sm[216] / TOT;
        float dice = 0.f;
        #pragma unroll
        for (int i = 0; i < 24; ++i) {
            const float I = sm[144 + i], G = sm[168 + i], P = sm[192 + i];
            dice += 1.0f - (2.0f * I + 1e-5f) / (G + P + 1e-5f);
        }
        dice *= (1.0f / 24.0f);
        const float ce_total = dice + ce;
        float pen = 0.f;
        #pragma unroll
        for (int i = 0; i < 144; ++i) pen += pen_mat[i] * sm[i];
        pen /= TOT;
        const float beta = powf(10.0f, floorf(log10f(ce_total)));
        const float g = (float)gstep[0], M = (float)miter[0];
        const float a1 = g / M, a0 = 1.0f - a1;
        out[0] = a1 * ce_total + a0 * beta * pen;
    }
}

extern "C" void kernel_launch(void* const* d_in, const int* in_sizes, int n_in,
                              void* d_out, int out_size, void* d_ws, size_t ws_size,
                              hipStream_t stream) {
    const float*     logits = (const float*)d_in[0];
    const long long* target = (const long long*)d_in[1];   // int64 per reference
    const float*     penm   = (const float*)d_in[2];
    const int*       gstep  = (const int*)d_in[3];         // low 32 bits valid for int32/int64
    const int*       miter  = (const int*)d_in[4];
    float* ws  = (float*)d_ws;
    float* out = (float*)d_out;

    domino_main_kernel<<<NBLK, 256, 0, stream>>>(logits, target, ws);
    domino_final_kernel<<<1, 256, 0, stream>>>(ws, penm, gstep, miter, out);
}
